// FeaturePropagation_2989297238647
// MI455X (gfx1250) — compile-verified, matrix operand order checked by reading, not run on hardware
//
#include <hip/hip_runtime.h>

typedef __bf16 bf16_t;
typedef __attribute__((ext_vector_type(16))) __bf16 v16bf;
typedef __attribute__((ext_vector_type(8)))  float  v8f;

#define B_CLOUDS 16
#define N_SRC    1024
#define N_DST    4096
#define C_IN     256
#define C_SKIP   128
#define C_HID    256
#define K_IN     (C_IN + C_SKIP)        // 384
#define M_TOTAL  (B_CLOUDS * N_DST)     // 65536

#define M_BLK 32
#define APAD  392    // 384+8 bf16 -> 784B row stride, 196 dwords (196%64==4, conflict-free)
#define BPAD  40     // 32+8  bf16 -> 80B stride, 20 dwords (gcd(20,64)=4 -> clean)
#define HPAD  264    // 256+8 bf16 -> 528B stride, 132 dwords (132%64==4)
#define EPSV  1e-16f

__global__ __launch_bounds__(256)
void fp_fused_kernel(const float* __restrict__ x,
                     const float* __restrict__ pos,
                     const float* __restrict__ x_skip,
                     const float* __restrict__ pos_skip,
                     const float* __restrict__ W1, const float* __restrict__ b1,
                     const float* __restrict__ W2, const float* __restrict__ b2,
                     float* __restrict__ out)
{
    // kNN scratch is dead once the bf16 A-tile is built -> overlay them.
    __shared__ union {
        bf16_t A[M_BLK * APAD];                       // 25088 B
        struct {
            float sx[N_SRC]; float sy[N_SRC]; float sz[N_SRC];  // 12288 B
            float cd[M_BLK * 8 * 3];                  // 3072 B
            int   ci[M_BLK * 8 * 3];                  // 3072 B
        } knn;
    } uu __attribute__((aligned(16)));
    __shared__ bf16_t sB[256 * BPAD] __attribute__((aligned(16)));   // 20480 B
    __shared__ bf16_t sH[M_BLK * HPAD] __attribute__((aligned(16))); // 16896 B
    __shared__ float  sw[M_BLK * 3];
    __shared__ int    si[M_BLK * 3];

    const int tid   = threadIdx.x;
    const int row0  = blockIdx.x * M_BLK;      // 4096 % 32 == 0 -> one cloud per block
    const int cloud = row0 / N_DST;

    // ---------------- Phase 1: stage source positions ----------------
    const float* posb = pos + (size_t)cloud * N_SRC * 3;
    for (int i = tid; i < N_SRC; i += 256) {
        uu.knn.sx[i] = posb[i * 3 + 0];
        uu.knn.sy[i] = posb[i * 3 + 1];
        uu.knn.sz[i] = posb[i * 3 + 2];
    }
    __syncthreads();

    // ---------------- Phase 2: 3-NN (8 threads per query row) --------
    {
        const int r   = tid >> 3;       // 0..31
        const int sub = tid & 7;        // 0..7, each scans 128 candidates
        const int p   = row0 + r;
        const float px = pos_skip[p * 3 + 0];
        const float py = pos_skip[p * 3 + 1];
        const float pz = pos_skip[p * 3 + 2];
        float d0 = 3.4e38f, d1 = 3.4e38f, d2v = 3.4e38f;
        int   i0 = 0, i1 = 0, i2 = 0;
        const int j0 = sub * (N_SRC / 8);
        for (int j = j0; j < j0 + N_SRC / 8; ++j) {
            const float dx = px - uu.knn.sx[j];
            const float dy = py - uu.knn.sy[j];
            const float dz = pz - uu.knn.sz[j];
            const float d  = dx * dx + dy * dy + dz * dz;
            if (d < d0)       { d2v = d1; i2 = i1; d1 = d0; i1 = i0; d0 = d; i0 = j; }
            else if (d < d1)  { d2v = d1; i2 = i1; d1 = d;  i1 = j; }
            else if (d < d2v) { d2v = d;  i2 = j; }
        }
        const int base = (r * 8 + sub) * 3;
        uu.knn.cd[base + 0] = d0;  uu.knn.cd[base + 1] = d1;  uu.knn.cd[base + 2] = d2v;
        uu.knn.ci[base + 0] = i0;  uu.knn.ci[base + 1] = i1;  uu.knn.ci[base + 2] = i2;
    }
    __syncthreads();

    if (tid < M_BLK) {
        float d0 = 3.4e38f, d1 = 3.4e38f, d2v = 3.4e38f;
        int   i0 = 0, i1 = 0, i2 = 0;
        for (int s = 0; s < 24; ++s) {
            const float d = uu.knn.cd[tid * 24 + s];
            const int   i = uu.knn.ci[tid * 24 + s];
            if (d < d0)       { d2v = d1; i2 = i1; d1 = d0; i1 = i0; d0 = d; i0 = i; }
            else if (d < d1)  { d2v = d1; i2 = i1; d1 = d;  i1 = i; }
            else if (d < d2v) { d2v = d;  i2 = i; }
        }
        const float w0 = 1.0f / (d0 + EPSV);
        const float w1 = 1.0f / (d1 + EPSV);
        const float w2 = 1.0f / (d2v + EPSV);
        const float inv = 1.0f / (w0 + w1 + w2);
        sw[tid * 3 + 0] = w0 * inv; sw[tid * 3 + 1] = w1 * inv; sw[tid * 3 + 2] = w2 * inv;
        si[tid * 3 + 0] = i0;       si[tid * 3 + 1] = i1;       si[tid * 3 + 2] = i2;
    }
    __syncthreads();

    // ---------------- Phase 3: build bf16 A-tile [32 x 384] ----------
    const float* xb = x + (size_t)cloud * N_SRC * C_IN;
    for (int r = 0; r < M_BLK; ++r) {
        const int p  = row0 + r;
        const int j0 = si[r * 3 + 0], j1 = si[r * 3 + 1], j2 = si[r * 3 + 2];
        const float w0 = sw[r * 3 + 0], w1 = sw[r * 3 + 1], w2 = sw[r * 3 + 2];
        {
            const int c = tid;  // 0..255, coalesced across the row
            const float v = w0 * xb[(size_t)j0 * C_IN + c]
                          + w1 * xb[(size_t)j1 * C_IN + c]
                          + w2 * xb[(size_t)j2 * C_IN + c];
            uu.A[r * APAD + c] = (bf16_t)v;
        }
        if (tid < C_SKIP)
            uu.A[r * APAD + C_IN + tid] = (bf16_t)x_skip[(size_t)p * C_SKIP + tid];
    }
    __syncthreads();

    // ---------------- Phase 4: GEMM1 (bf16 WMMA, f32 acc) ------------
    const int lane   = tid & 31;
    const int wid    = tid >> 5;
    const int wave_m = wid >> 2;            // 0..1 -> 16-row tile
    const int wave_n = wid & 3;             // 0..3 -> 64-col slab
    const int lm     = lane & 15;
    const int kb     = (lane >> 4) * 8;     // 0 or 8 (16-bit A/B fragment layout)
    const int mA     = wave_m * 16 + lm;

    union Frag { v16bf v; uint4 q[2]; };
    v8f acc[4] = {};
    const v8f vzero = {0.f, 0.f, 0.f, 0.f, 0.f, 0.f, 0.f, 0.f};

    for (int ks = 0; ks < K_IN / 32; ++ks) {
        const int k0 = ks * 32;
        // transpose-stage W1 K-slice into LDS as [N][K] bf16 (coalesced reads)
        #pragma unroll
        for (int kk = 0; kk < 32; ++kk)
            sB[tid * BPAD + kk] = (bf16_t)W1[(size_t)(k0 + kk) * C_HID + tid];
        __syncthreads();

        Frag a;
        const bf16_t* pa = &uu.A[mA * APAD + k0 + kb];
        a.q[0] = *reinterpret_cast<const uint4*>(pa);        // K = kb..kb+7
        a.q[1] = *reinterpret_cast<const uint4*>(pa + 16);   // K = kb+16..kb+23

        #pragma unroll
        for (int nj = 0; nj < 4; ++nj) {
            Frag bfg;
            const bf16_t* pb = &sB[(wave_n * 64 + nj * 16 + lm) * BPAD + kb];
            bfg.q[0] = *reinterpret_cast<const uint4*>(pb);
            bfg.q[1] = *reinterpret_cast<const uint4*>(pb + 16);
            acc[nj] = __builtin_amdgcn_wmma_f32_16x16x32_bf16(
                false, a.v, false, bfg.v, (short)0, acc[nj], false, false);
        }
        __syncthreads();
    }

    // ---------------- Phase 5: bias + ReLU -> bf16 hidden tile -------
    const int rowH = wave_m * 16 + 8 * (lane >> 4);   // D layout: lane>=16 holds M=r+8
    #pragma unroll
    for (int nj = 0; nj < 4; ++nj) {
        const int n = wave_n * 64 + nj * 16 + lm;
        const float bias = b1[n];
        #pragma unroll
        for (int r = 0; r < 8; ++r) {
            float h = acc[nj][r] + bias;
            h = h > 0.0f ? h : 0.0f;
            sH[(rowH + r) * HPAD + n] = (bf16_t)h;
        }
        acc[nj] = vzero;
    }
    __syncthreads();

    // ---------------- Phase 6: GEMM2 ---------------------------------
    for (int ks = 0; ks < C_HID / 32; ++ks) {
        const int k0 = ks * 32;
        #pragma unroll
        for (int kk = 0; kk < 32; ++kk)
            sB[tid * BPAD + kk] = (bf16_t)W2[(size_t)(k0 + kk) * C_HID + tid];
        __syncthreads();

        Frag a;
        const bf16_t* pa = &sH[mA * HPAD + k0 + kb];
        a.q[0] = *reinterpret_cast<const uint4*>(pa);
        a.q[1] = *reinterpret_cast<const uint4*>(pa + 16);

        #pragma unroll
        for (int nj = 0; nj < 4; ++nj) {
            Frag bfg;
            const bf16_t* pb = &sB[(wave_n * 64 + nj * 16 + lm) * BPAD + kb];
            bfg.q[0] = *reinterpret_cast<const uint4*>(pb);
            bfg.q[1] = *reinterpret_cast<const uint4*>(pb + 16);
            acc[nj] = __builtin_amdgcn_wmma_f32_16x16x32_bf16(
                false, a.v, false, bfg.v, (short)0, acc[nj], false, false);
        }
        __syncthreads();
    }

    // ---------------- Phase 7: bias + store --------------------------
    #pragma unroll
    for (int nj = 0; nj < 4; ++nj) {
        const int n = wave_n * 64 + nj * 16 + lm;
        const float bias = b2[n];
        #pragma unroll
        for (int r = 0; r < 8; ++r) {
            const int m = rowH + r;
            out[(size_t)(row0 + m) * C_HID + n] = acc[nj][r] + bias;
        }
    }
}

// Tuple tail: pos_skip passthrough + batch_skip values.
__global__ __launch_bounds__(256)
void fp_tail_kernel(const float* __restrict__ pos_skip,
                    float* __restrict__ out_tail, int tail_elems)
{
    const int i = blockIdx.x * 256 + threadIdx.x;
    if (i < M_TOTAL * 3) out_tail[i] = pos_skip[i];
    if (i < M_TOTAL && tail_elems > 0) {
        if (tail_elems >= 2 * M_TOTAL) {   // harness counted int64 as 2 slots
            long long* bt = reinterpret_cast<long long*>(out_tail + M_TOTAL * 3);
            bt[i] = (long long)(i / N_DST);
        } else {
            float* bt = out_tail + M_TOTAL * 3;
            bt[i] = (float)(i / N_DST);
        }
    }
}

extern "C" void kernel_launch(void* const* d_in, const int* in_sizes, int n_in,
                              void* d_out, int out_size, void* d_ws, size_t ws_size,
                              hipStream_t stream) {
    (void)in_sizes; (void)n_in; (void)d_ws; (void)ws_size;
    const float* x        = (const float*)d_in[0];
    const float* pos      = (const float*)d_in[1];
    const float* x_skip   = (const float*)d_in[3];
    const float* pos_skip = (const float*)d_in[4];
    const float* W1       = (const float*)d_in[6];
    const float* b1       = (const float*)d_in[7];
    const float* W2       = (const float*)d_in[8];
    const float* b2       = (const float*)d_in[9];
    float* out = (float*)d_out;

    fp_fused_kernel<<<M_TOTAL / M_BLK, 256, 0, stream>>>(
        x, pos, x_skip, pos_skip, W1, b1, W2, b2, out);

    const int tail_elems = out_size - M_TOTAL * C_HID - M_TOTAL * 3;
    fp_tail_kernel<<<(M_TOTAL * 3 + 255) / 256, 256, 0, stream>>>(
        pos_skip, out + (size_t)M_TOTAL * C_HID, tail_elems);
}
